// EnergyAdaptor_79559974191188
// MI455X (gfx1250) — matrix operand order repaired and checked by reading, not compiled, not run on hardware
//
#include <hip/hip_runtime.h>
#include <hip/hip_bf16.h>
#include <stdint.h>

// ---------------- problem constants (match reference) ----------------
#define B_SZ   32
#define T_LEN  1024      // T_SRC
#define C_IN1  512
#define CHN    256       // CH
#define T_DE   4096
#define KTAP   5
#define LEAKYF 0.3f
#define EPSV   1e-5f

typedef __attribute__((ext_vector_type(16))) __bf16 v16bf;
typedef __attribute__((ext_vector_type(8)))  float  v8f;

__device__ __forceinline__ unsigned short f2bf(float f) {
  // round-to-nearest-even f32 -> bf16
  unsigned int u = __float_as_uint(f);
  u += 0x7FFFu + ((u >> 16) & 1u);
  return (unsigned short)(u >> 16);
}

// ---------------------------------------------------------------------
// Weight repack: w[co][ci][tap] (f32) -> bf16 fragments laid out so each
// WMMA B-fragment (lane g,n holds K = g*16+0..15 of column n) is one
// contiguous 32-byte load. Flat order: tap, kt, nt, lane, e.
// ---------------------------------------------------------------------
__global__ __launch_bounds__(256) void repack_w(const float* __restrict__ w,
                                                unsigned short* __restrict__ wp,
                                                int CIN, int KT, int total) {
  int idx = blockIdx.x * 256 + threadIdx.x;
  if (idx >= total) return;
  int e    = idx & 15;
  int r    = idx >> 4;
  int lane = r & 31;  r >>= 5;
  int nt   = r & 15;  r >>= 4;
  int kt   = r % KT;
  int t    = r / KT;
  int g  = lane >> 4, n = lane & 15;
  int co = nt * 16 + n;
  int ci = kt * 32 + g * 16 + e;
  wp[idx] = f2bf(w[(co * CIN + ci) * KTAP + t]);
}

// ---------------------------------------------------------------------
// Conv1d-as-GEMM with bf16 WMMA. Block = 256 thr = 8 waves -> 64(M)x32(N)
// output tile for one batch. LDS stages a 68-row (±2 halo) x 32-K bf16
// activation tile per K-chunk; 5 taps are 5 accumulating WMMAs with a
// shifted A row base. Weights come pre-fragmented from wpack (L2-resident).
// ---------------------------------------------------------------------
template<int CIN, bool AF32>
__global__ __launch_bounds__(256) void conv_wmma(const float* __restrict__ Af,
                                                 const unsigned short* __restrict__ Ah,
                                                 const unsigned short* __restrict__ wpack,
                                                 float* __restrict__ Y) {
  constexpr int KT = CIN / 32;
  const int tid  = threadIdx.x;
  const int lane = tid & 31;
  const int wave = tid >> 5;
  const int mw   = wave & 3;           // M sub-tile 0..3
  const int nw   = wave >> 2;          // N sub-tile 0..1
  const int bx   = blockIdx.x;         // 0 .. B*(T/64)-1
  const int b    = bx >> 4;            // T/64 == 16
  const int t0   = (bx & 15) << 6;
  const int ntg  = blockIdx.y * 2 + nw; // 16-col tile index 0..15

  __shared__ unsigned short sA[68 * 40];   // 68 halo rows, stride 40 (pad)

  const int g    = lane >> 4;
  const int rowm = lane & 15;
  v8f acc = {0.f, 0.f, 0.f, 0.f, 0.f, 0.f, 0.f, 0.f};

  for (int kc = 0; kc < KT; ++kc) {
    __syncthreads();
    for (int idx = tid; idx < 68 * 32; idx += 256) {
      int hr = idx >> 5, col = idx & 31;
      int tt = t0 + hr - 2;
      unsigned short v = 0;
      if (tt >= 0 && tt < T_LEN) {
        int src = (b * T_LEN + tt) * CIN + kc * 32 + col;
        if (AF32) v = f2bf(Af[src]);
        else      v = Ah[src];
      }
      sA[hr * 40 + col] = v;
    }
    __syncthreads();
#pragma unroll
    for (int tap = 0; tap < KTAP; ++tap) {
      // A fragment: lane half g holds K = g*16+0..15 (contiguous, matches B)
      int ra = mw * 16 + tap + rowm;           // halo row
      v16bf a, bb;
      uint4* ap = reinterpret_cast<uint4*>(&a);
      ap[0] = *reinterpret_cast<const uint4*>(&sA[ra * 40 + g * 16]);
      ap[1] = *reinterpret_cast<const uint4*>(&sA[ra * 40 + g * 16 + 8]);
      // B fragment: one contiguous 32B chunk per lane from packed weights
      const unsigned short* wp =
          wpack + ((size_t)(((tap * KT + kc) * 16 + ntg) * 32 + lane) << 4);
      uint4* bp = reinterpret_cast<uint4*>(&bb);
      bp[0] = reinterpret_cast<const uint4*>(wp)[0];
      bp[1] = reinterpret_cast<const uint4*>(wp)[1];

      acc = __builtin_amdgcn_wmma_f32_16x16x32_bf16(
          /*neg_a=*/false, a, /*neg_b=*/false, bb,
          /*c_mod=*/(short)0, acc, /*reuse_a=*/false, /*reuse_b=*/false);
    }
  }

  // C/D layout: VGPR r -> M = r + (lane/16)*8, N = lane%16
  const int mlo = g * 8;
#pragma unroll
  for (int r = 0; r < 8; ++r) {
    int row = t0 + mw * 16 + mlo + r;
    Y[(size_t)(b * T_LEN + row) * CHN + ntg * 16 + rowm] = acc[r];
  }
}

// ---------------------------------------------------------------------
// bias + leaky-relu + LayerNorm(256) -> bf16 h1.  One wave per row.
// ---------------------------------------------------------------------
__global__ __launch_bounds__(256) void ln_act_kernel(const float* __restrict__ Y,
                                                     const float* __restrict__ bias,
                                                     const float* __restrict__ gam,
                                                     const float* __restrict__ bet,
                                                     unsigned short* __restrict__ H) {
  int row  = blockIdx.x * 8 + (threadIdx.x >> 5);
  int lane = threadIdx.x & 31;
  float v[8], s = 0.f, q = 0.f;
#pragma unroll
  for (int i = 0; i < 8; ++i) {
    int c = i * 32 + lane;
    float x = Y[(size_t)row * CHN + c] + bias[c];
    x = x > 0.f ? x : LEAKYF * x;
    v[i] = x; s += x; q += x * x;
  }
  for (int off = 16; off; off >>= 1) {
    s += __shfl_xor(s, off, 32);
    q += __shfl_xor(q, off, 32);
  }
  float m   = s * (1.f / CHN);
  float var = q * (1.f / CHN) - m * m;
  float inv = rsqrtf(var + EPSV);
#pragma unroll
  for (int i = 0; i < 8; ++i) {
    int c = i * 32 + lane;
    H[(size_t)row * CHN + c] = f2bf((v[i] - m) * inv * gam[c] + bet[c]);
  }
}

// ---------------------------------------------------------------------
// bias + leaky + LayerNorm + (h @ w_lin + b_lin), masked -> energy_pred.
// ---------------------------------------------------------------------
__global__ __launch_bounds__(256) void ln_lin_kernel(const float* __restrict__ Y,
                                                     const float* __restrict__ bias,
                                                     const float* __restrict__ gam,
                                                     const float* __restrict__ bet,
                                                     const float* __restrict__ wl,
                                                     const float* __restrict__ bl,
                                                     const unsigned char* __restrict__ mask,
                                                     float* __restrict__ out) {
  int row  = blockIdx.x * 8 + (threadIdx.x >> 5);
  int lane = threadIdx.x & 31;
  float v[8], s = 0.f, q = 0.f;
#pragma unroll
  for (int i = 0; i < 8; ++i) {
    int c = i * 32 + lane;
    float x = Y[(size_t)row * CHN + c] + bias[c];
    x = x > 0.f ? x : LEAKYF * x;
    v[i] = x; s += x; q += x * x;
  }
  for (int off = 16; off; off >>= 1) {
    s += __shfl_xor(s, off, 32);
    q += __shfl_xor(q, off, 32);
  }
  float m   = s * (1.f / CHN);
  float var = q * (1.f / CHN) - m * m;
  float inv = rsqrtf(var + EPSV);
  float acc = 0.f;
#pragma unroll
  for (int i = 0; i < 8; ++i) {
    int c = i * 32 + lane;
    acc += ((v[i] - m) * inv * gam[c] + bet[c]) * wl[c];
  }
  for (int off = 16; off; off >>= 1) acc += __shfl_xor(acc, off, 32);
  if (lane == 0) out[row] = mask[row] ? 0.f : (acc + bl[0]);
}

// ---------------------------------------------------------------------
// Average target over durations. One block per batch row: prefix sums of
// values / nonzero counts (4096) and durations (1024) in LDS.
// ---------------------------------------------------------------------
__global__ __launch_bounds__(256) void avg_dur_kernel(const float* __restrict__ target,
                                                      const int* __restrict__ dr,
                                                      float* __restrict__ avg) {
  const int b = blockIdx.x, tid = threadIdx.x;
  __shared__ float vcum[T_DE + 1];
  __shared__ float ncum[T_DE + 1];
  __shared__ int   dcum[T_LEN + 1];
  __shared__ float pv[256], pn[256];
  __shared__ int   pd[256];
  const float* tg = target + (size_t)b * T_DE;

  {
    int base = tid * 16;
    float s = 0.f, nz = 0.f;
    for (int j = 0; j < 16; ++j) {
      float x = tg[base + j];
      s += x; nz += (x != 0.f) ? 1.f : 0.f;
    }
    pv[tid] = s; pn[tid] = nz;
  }
  {
    int base = tid * 4;
    int s = 0;
    for (int j = 0; j < 4; ++j) s += dr[b * T_LEN + base + j];
    pd[tid] = s;
  }
  __syncthreads();
  if (tid == 0) {
    float rs = 0.f, rn = 0.f; int rd = 0;
    for (int i = 0; i < 256; ++i) {
      float t1 = pv[i]; pv[i] = rs; rs += t1;
      float t2 = pn[i]; pn[i] = rn; rn += t2;
      int   t3 = pd[i]; pd[i] = rd; rd += t3;
    }
    vcum[0] = 0.f; ncum[0] = 0.f; dcum[0] = 0;
  }
  __syncthreads();
  {
    int base = tid * 16;
    float rs = pv[tid], rn = pn[tid];
    for (int j = 0; j < 16; ++j) {
      float x = tg[base + j];
      rs += x; rn += (x != 0.f) ? 1.f : 0.f;
      vcum[base + j + 1] = rs; ncum[base + j + 1] = rn;
    }
  }
  {
    int base = tid * 4;
    int rd = pd[tid];
    for (int j = 0; j < 4; ++j) {
      rd += dr[b * T_LEN + base + j];
      dcum[base + j + 1] = rd;
    }
  }
  __syncthreads();
  for (int i = tid; i < T_LEN; i += 256) {
    int s = dcum[i], e = dcum[i + 1];     // starts / ends
    float sum = vcum[e] - vcum[s];
    float nel = ncum[e] - ncum[s];
    avg[b * T_LEN + i] = (nel == 0.f) ? 0.f : sum / nel;
  }
}

// ---------------------------------------------------------------------
// energy_emb = conv1d(avg, w_emb[256][1][3], pad 1) -> (B, CH, T)
// ---------------------------------------------------------------------
__global__ __launch_bounds__(256) void emb_kernel(const float* __restrict__ avg,
                                                  const float* __restrict__ wemb,
                                                  const float* __restrict__ bemb,
                                                  float* __restrict__ out) {
  int idx = blockIdx.x * 256 + threadIdx.x;   // b*CH*T threads
  int t  = idx & (T_LEN - 1);
  int co = (idx >> 10) & (CHN - 1);
  int b  = idx >> 18;
  const float* a = avg + (size_t)b * T_LEN;
  float w0 = wemb[co * 3 + 0], w1 = wemb[co * 3 + 1], w2 = wemb[co * 3 + 2];
  float v = bemb[co] + w1 * a[t];
  if (t > 0)          v += w0 * a[t - 1];
  if (t < T_LEN - 1)  v += w2 * a[t + 1];
  out[idx] = v;
}

// ---------------------------------------------------------------------
extern "C" void kernel_launch(void* const* d_in, const int* in_sizes, int n_in,
                              void* d_out, int out_size, void* d_ws, size_t ws_size,
                              hipStream_t stream) {
  const float*         x      = (const float*)d_in[0];
  const float*         target = (const float*)d_in[1];
  const int*           dr     = (const int*)d_in[2];
  const unsigned char* mask   = (const unsigned char*)d_in[3];
  const float* w1  = (const float*)d_in[4];
  const float* b1  = (const float*)d_in[5];
  const float* g1  = (const float*)d_in[6];
  const float* be1 = (const float*)d_in[7];
  const float* w2  = (const float*)d_in[8];
  const float* b2  = (const float*)d_in[9];
  const float* g2  = (const float*)d_in[10];
  const float* be2 = (const float*)d_in[11];
  const float* wl  = (const float*)d_in[12];
  const float* bl  = (const float*)d_in[13];
  const float* wem = (const float*)d_in[14];
  const float* bem = (const float*)d_in[15];

  char* ws = (char*)d_ws;
  size_t off = 0;
  unsigned short* wp1 = (unsigned short*)(ws + off); off += (size_t)KTAP * C_IN1 * CHN * 2;
  unsigned short* wp2 = (unsigned short*)(ws + off); off += (size_t)KTAP * CHN * CHN * 2;
  off = (off + 255) & ~(size_t)255;
  float*          Ybuf = (float*)(ws + off);          off += (size_t)B_SZ * T_LEN * CHN * 4;
  unsigned short* H1   = (unsigned short*)(ws + off); off += (size_t)B_SZ * T_LEN * CHN * 2;

  float* out_pred = (float*)d_out;
  float* out_avg  = out_pred + B_SZ * T_LEN;
  float* out_emb  = out_avg + B_SZ * T_LEN;

  // weight fragment repack (tiny)
  repack_w<<<(KTAP * C_IN1 * CHN + 255) / 256, 256, 0, stream>>>(
      w1, wp1, C_IN1, C_IN1 / 32, KTAP * C_IN1 * CHN);
  repack_w<<<(KTAP * CHN * CHN + 255) / 256, 256, 0, stream>>>(
      w2, wp2, CHN, CHN / 32, KTAP * CHN * CHN);

  dim3 cg(B_SZ * (T_LEN / 64), CHN / 32);
  // conv1 (f32 x -> bf16 frags -> f32 acc)
  conv_wmma<C_IN1, true><<<cg, 256, 0, stream>>>(x, nullptr, wp1, Ybuf);
  // bias+leaky+LN -> bf16 h1
  ln_act_kernel<<<B_SZ * T_LEN / 8, 256, 0, stream>>>(Ybuf, b1, g1, be1, H1);
  // conv2 (bf16 h1)
  conv_wmma<CHN, false><<<cg, 256, 0, stream>>>(nullptr, H1, wp2, Ybuf);
  // bias+leaky+LN + linear head -> energy_pred
  ln_lin_kernel<<<B_SZ * T_LEN / 8, 256, 0, stream>>>(Ybuf, b2, g2, be2, wl, bl, mask, out_pred);
  // duration averaging -> avg_energy_target
  avg_dur_kernel<<<B_SZ, 256, 0, stream>>>(target, dr, out_avg);
  // K=3 embedding conv -> energy_emb
  emb_kernel<<<(B_SZ * CHN * T_LEN) / 256, 256, 0, stream>>>(out_avg, wem, bem, out_emb);
}
